// FlamingoPerceiverResampler_9174050144687
// MI455X (gfx1250) — matrix operand order repaired and checked by reading, not compiled, not run on hardware
//
#include <hip/hip_runtime.h>

// ---------------- types ----------------
typedef __attribute__((ext_vector_type(16))) __bf16 v16bf;
typedef __attribute__((ext_vector_type(8)))  __bf16 v8bf;
typedef __attribute__((ext_vector_type(4)))  __bf16 v4bf;
typedef __attribute__((ext_vector_type(8)))  float  v8f;
typedef __attribute__((ext_vector_type(4)))  float  v4f;
typedef __attribute__((ext_vector_type(4)))  int    v4i;
typedef __attribute__((ext_vector_type(4)))  unsigned int u32x4;
typedef __attribute__((ext_vector_type(8)))  int    i32x8;
typedef __attribute__((ext_vector_type(4)))  int    i32x4;

union FragU {
  v8bf   c[2];
  __bf16 u[16];
  v16bf  v;
};

#define DEV __device__ __forceinline__

// ---------------- CDNA5 capability probes ----------------
#if __has_builtin(__builtin_amdgcn_global_load_async_to_lds_b128) && \
    __has_builtin(__builtin_amdgcn_s_wait_asynccnt)
#define HAVE_ASYNC 1
#else
#define HAVE_ASYNC 0
#endif

#if __has_builtin(__builtin_amdgcn_tensor_load_to_lds) && \
    __has_builtin(__builtin_amdgcn_s_wait_tensorcnt)
#define HAVE_TDM 1
#else
#define HAVE_TDM 0
#endif

#if HAVE_ASYNC
// 16-byte global -> LDS async copy (one per lane), tracked by ASYNCcnt
DEV void async_cp16(const void* g, void* l) {
  __builtin_amdgcn_global_load_async_to_lds_b128(
      (__attribute__((address_space(1))) v4i*)g,
      (__attribute__((address_space(3))) v4i*)l, 0, 0);
}
#endif

DEV unsigned lds_addr_of(const void* p) {
  return (unsigned)(size_t)(__attribute__((address_space(3))) const void*)p;
}

#if HAVE_TDM
// TDM 2-D tile load: tileRows x 32 bf16 elements, global row stride
// strideElems, into LDS rows padded to 40 bf16 (pad 4 DWORDs every 64B).
// Descriptor per CDNA5 ISA D# groups 0/1 (type=2, count=1, data_size=2B).
// This toolchain's builtin is the 6-arg form:
//   (u32x4 g0, i32x8 g1, i32x4 g2, i32x4 g3, i32x8, i32 cpol)
DEV void tdm_load_2d(const void* gsrc, unsigned ldsAddr, int tileRows,
                     int strideElems) {
  const unsigned long long ga = (unsigned long long)(size_t)gsrc;
  u32x4 g0;
  g0[0] = 1u;                                   // count=1, user mode
  g0[1] = ldsAddr;                              // lds_addr (bytes)
  g0[2] = (unsigned)ga;                         // global_addr[31:0]
  g0[3] = (unsigned)((ga >> 32) & 0x01ffffffu)  // global_addr[56:32]
          | (2u << 30);                         // type=2 ("image")
  i32x8 g1;
  // data_size=1 (2B) | pad_enable | pad_interval=3 (64B) | pad_amount=3 (4 dw)
  g1[0] = (1 << 16) | (1 << 20) | (3 << 22) | (3 << 25);
  g1[1] = (strideElems & 0xffff) << 16;         // tensor_dim0[15:0]
  g1[2] = ((strideElems >> 16) & 0xffff)        // tensor_dim0[31:16]
          | ((tileRows & 0xffff) << 16);        // tensor_dim1[15:0]
  g1[3] = (32 << 16);                           // tensor_dim1[31:16]=0 | tile_dim0=32
  g1[4] = tileRows & 0xffff;                    // tile_dim1 | tile_dim2=0
  g1[5] = strideElems;                          // tensor_dim0_stride[31:0]
  g1[6] = 0;                                    // stride hi | dim1_stride lo
  g1[7] = 0;
  const i32x4 gz4 = {0, 0, 0, 0};
  const i32x8 gz8 = {0, 0, 0, 0, 0, 0, 0, 0};
  __builtin_amdgcn_tensor_load_to_lds(g0, g1, gz4, gz4, gz8, 0);
}
#endif

// ---------------- problem constants ----------------
constexpr int Dm    = 1024;
constexpr int DH    = 64;
constexpr int Hh    = 8;
constexpr int INNER = 512;    // DH*H
constexpr int Lnum  = 6;
constexpr int NLAT  = 64;
constexpr int FFD   = 4096;
constexpr int Tt    = 8;
constexpr int BT    = 32;     // B*T
constexpr int MEDIA = 1024;   // F*V
constexpr int TOK   = 1088;   // MEDIA + NLAT
constexpr int KV2   = 1024;   // 2*INNER

// =====================================================================
// WMMA GEMM: C[64-tile, 128-tile] = A_bf16[M,K] @ Wt_bf16[N,K]^T
// Weights PRE-TRANSPOSED [N][K]; tiles staged by the Tensor Data Mover
// (one 2-D descriptor per tile per K-step, LDS-padded rows), falling back
// to async global->LDS b128, then to plain loads. Double-buffered.
// EPI: 0 = bf16 store (scale, optional bias[col])
//      1 = fp32 store + residual
//      2 = exact GELU -> bf16
//      3 = split KV: col<INNER -> K row-major; col>=INNER -> V transposed
// =====================================================================
template <int EPI>
__global__ __launch_bounds__(256) void gemm_bf16_wmma(
    const __bf16* __restrict__ A, int lda, size_t aBatch,
    const __bf16* __restrict__ Wt, int ldk,
    const float* __restrict__ bias,
    void* __restrict__ Cv, void* __restrict__ Cv2,
    int ldc, size_t cBatch, int cRowOff,
    const float* __restrict__ R, int ldr, size_t rBatch,
    int K, float scale)
{
  __shared__ __bf16 As[2][64][40];    // [buf][m][k]  (40: +4dw TDM pad)
  __shared__ __bf16 Bs[2][128][40];   // [buf][n][k]

  const int bz = blockIdx.z;
  const int m0 = blockIdx.y * 64;
  const int n0 = blockIdx.x * 128;
  A += (size_t)bz * aBatch;

  const int tid  = threadIdx.x;
  const int wave = __builtin_amdgcn_readfirstlane(tid >> 5);
  const int lane = tid & 31;
  const int mb   = wave & 3;    // m sub-tile 0..3
  const int ng   = wave >> 2;   // n half 0..1 (4 subtiles of 16)

  const v8f z = {0.f, 0.f, 0.f, 0.f, 0.f, 0.f, 0.f, 0.f};
  v8f acc[4] = {z, z, z, z};

  const int lr = tid >> 2;          // A stage: row 0..63
  const int lc = (tid & 3) * 8;     // A stage: col 0,8,16,24
  const int bn = tid >> 1;          // B stage: row 0..127
  const int bc = (tid & 1) * 16;    // B stage: col 0 or 16

  v8bf aR{}, bR0{}, bR1{};

  auto stageNext = [&](int kc, int buf) {
#if HAVE_TDM
    if (wave == 0)
      tdm_load_2d(A + (size_t)m0 * lda + kc, lds_addr_of(&As[buf][0][0]), 64,
                  lda);
    else if (wave == 1)
      tdm_load_2d(Wt + (size_t)n0 * ldk + kc, lds_addr_of(&Bs[buf][0][0]), 128,
                  ldk);
#elif HAVE_ASYNC
    async_cp16(A + (size_t)(m0 + lr) * lda + kc + lc, &As[buf][lr][lc]);
    const __bf16* wsrc = Wt + (size_t)(n0 + bn) * ldk + kc + bc;
    async_cp16(wsrc, &Bs[buf][bn][bc]);
    async_cp16(wsrc + 8, &Bs[buf][bn][bc + 8]);
#else
    aR  = *reinterpret_cast<const v8bf*>(A + (size_t)(m0 + lr) * lda + kc + lc);
    const __bf16* wsrc = Wt + (size_t)(n0 + bn) * ldk + kc + bc;
    bR0 = *reinterpret_cast<const v8bf*>(wsrc);
    bR1 = *reinterpret_cast<const v8bf*>(wsrc + 8);
    (void)buf;
#endif
  };
  auto commit = [&](int buf) {
#if !HAVE_TDM && !HAVE_ASYNC
    *reinterpret_cast<v8bf*>(&As[buf][lr][lc])     = aR;
    *reinterpret_cast<v8bf*>(&Bs[buf][bn][bc])     = bR0;
    *reinterpret_cast<v8bf*>(&Bs[buf][bn][bc + 8]) = bR1;
#else
    (void)buf;
#endif
  };
  auto waitStage = [&]() {
#if HAVE_TDM
    __builtin_amdgcn_s_wait_tensorcnt(0);
#elif HAVE_ASYNC
    __builtin_amdgcn_s_wait_asynccnt(0);
#endif
  };

  const int nIt = K >> 5;

  stageNext(0, 0);
  commit(0);
  waitStage();
  __syncthreads();

  for (int it = 0; it < nIt; ++it) {
    const int  cur  = it & 1;
    const int  nxt  = cur ^ 1;
    const bool more = (it + 1) < nIt;

    if (more) stageNext((it + 1) * 32, nxt);   // DMA overlaps WMMA

    FragU a;
    const int arow = mb * 16 + (lane & 15);
    const int aoff = (lane < 16) ? 0 : 16;
    a.c[0] = *reinterpret_cast<const v8bf*>(&As[cur][arow][aoff]);
    a.c[1] = *reinterpret_cast<const v8bf*>(&As[cur][arow][aoff + 8]);
    const int koff = (lane < 16) ? 0 : 16;
#pragma unroll
    for (int j = 0; j < 4; ++j) {
      FragU b;
      const int bnr = ng * 64 + j * 16 + (lane & 15);
      b.c[0] = *reinterpret_cast<const v8bf*>(&Bs[cur][bnr][koff]);
      b.c[1] = *reinterpret_cast<const v8bf*>(&Bs[cur][bnr][koff + 8]);
      acc[j] = __builtin_amdgcn_wmma_f32_16x16x32_bf16(
          false, a.v, false, b.v, (short)0, acc[j], false, false);
    }

    if (more) commit(nxt);
    waitStage();
    __syncthreads();
  }

  // epilogue
  const int rowbase = m0 + mb * 16 + ((lane < 16) ? 0 : 8);
#pragma unroll
  for (int j = 0; j < 4; ++j) {
    const int col = n0 + ng * 64 + j * 16 + (lane & 15);
#pragma unroll
    for (int r = 0; r < 8; ++r) {
      const int row = rowbase + r;
      if constexpr (EPI == 0) {
        float v = acc[j][r] * scale;
        if (bias) v += bias[col];
        ((__bf16*)Cv)[(size_t)bz * cBatch + (size_t)(cRowOff + row) * ldc +
                      col] = (__bf16)v;
      } else if constexpr (EPI == 1) {
        const float res = R[(size_t)bz * rBatch + (size_t)row * ldr + col];
        ((float*)Cv)[(size_t)bz * cBatch + (size_t)(cRowOff + row) * ldc +
                     col] = acc[j][r] + res;
      } else if constexpr (EPI == 2) {
        const float x = acc[j][r];
        const float g = 0.5f * x * (1.0f + erff(x * 0.70710678118f));
        ((__bf16*)Cv)[(size_t)bz * cBatch + (size_t)(cRowOff + row) * ldc +
                      col] = (__bf16)g;
      } else {  // EPI == 3: split K / V-transposed
        float v = acc[j][r];
        if (bias) v += bias[col];
        if (col < INNER) {
          ((__bf16*)Cv)[(size_t)bz * cBatch + (size_t)(cRowOff + row) * INNER +
                        col] = (__bf16)v;
        } else {
          ((__bf16*)Cv2)[(size_t)bz * cBatch + (size_t)(col - INNER) * TOK +
                         (cRowOff + row)] = (__bf16)v;
        }
      }
    }
  }
}

// =====================================================================
// fp32[K][N] -> bf16[N][K]  (transpose + convert; one-time per weight)
// =====================================================================
__global__ __launch_bounds__(256) void cvt_bf16_t(
    const float* __restrict__ in, __bf16* __restrict__ out, int K, int N)
{
  const int idx = blockIdx.x * blockDim.x + threadIdx.x;  // N*K/4 threads
  const int kdiv = K >> 2;
  if (idx >= kdiv * N) return;
  const int n  = idx / kdiv;
  const int kq = (idx - n * kdiv) * 4;
  __bf16* o = out + (size_t)n * K + kq;
#pragma unroll
  for (int i = 0; i < 4; ++i) o[i] = (__bf16)in[(size_t)(kq + i) * N + n];
}

// folded media Wkv, transposed: out[n][k] = g[k]*Wkv[k][n]
__global__ __launch_bounds__(256) void fold_wkv_t(
    const float* __restrict__ Wkv, const float* __restrict__ g,
    __bf16* __restrict__ out)
{
  const int idx = blockIdx.x * blockDim.x + threadIdx.x;  // KV2*Dm/4
  const int kdiv = Dm >> 2;
  const int n  = idx / kdiv;
  const int kq = (idx - n * kdiv) * 4;
  __bf16* o = out + (size_t)n * Dm + kq;
#pragma unroll
  for (int i = 0; i < 4; ++i)
    o[i] = (__bf16)(g[kq + i] * Wkv[(size_t)(kq + i) * KV2 + n]);
}

// bias[n] = sum_k b[k] * Wkv[k][n]   (rank-1 LN-beta fold)
__global__ __launch_bounds__(256) void bias_wkv(
    const float* __restrict__ Wkv, const float* __restrict__ b,
    float* __restrict__ bias)
{
  const int n = blockIdx.x * blockDim.x + threadIdx.x;  // 0..1023
  float s = 0.f;
  for (int k = 0; k < Dm; ++k) s += b[k] * Wkv[(size_t)k * KV2 + n];
  bias[n] = s;
}

// =====================================================================
// prep: x + frame_emb, parameter-free LN over D -> bf16 base
// =====================================================================
__global__ __launch_bounds__(256) void prep_media(
    const float* __restrict__ x, const float* __restrict__ femb,
    __bf16* __restrict__ out)
{
  const int row = blockIdx.x;          // 0 .. BT*MEDIA-1
  const int bt  = row >> 10;
  const int t   = bt & (Tt - 1);
  const int tid = threadIdx.x;
  const float* xr = x + (size_t)row * Dm;
  const float* fr = femb + (size_t)t * Dm;

  float v[4], s = 0.f, s2 = 0.f;
#pragma unroll
  for (int i = 0; i < 4; ++i) {
    const int d = tid + i * 256;
    v[i] = xr[d] + fr[d];
    s += v[i];
    s2 += v[i] * v[i];
  }
  __shared__ float r1[256], r2[256];
  r1[tid] = s; r2[tid] = s2;
  __syncthreads();
  for (int off = 128; off > 0; off >>= 1) {
    if (tid < off) { r1[tid] += r1[tid + off]; r2[tid] += r2[tid + off]; }
    __syncthreads();
  }
  const float mean = r1[0] * (1.0f / Dm);
  const float var  = r2[0] * (1.0f / Dm) - mean * mean;
  const float rstd = rsqrtf(var + 1e-5f);
  __bf16* orow = out + (size_t)row * Dm;
#pragma unroll
  for (int i = 0; i < 4; ++i) {
    const int d = tid + i * 256;
    orow[d] = (__bf16)((v[i] - mean) * rstd);
  }
}

// =====================================================================
// row LayerNorm with affine; OBF: bf16 out else fp32
// =====================================================================
template <bool OBF>
__global__ __launch_bounds__(256) void ln_rows(
    const float* __restrict__ in, const float* __restrict__ g,
    const float* __restrict__ b, void* __restrict__ out)
{
  const int row = blockIdx.x;
  const int tid = threadIdx.x;
  const float* xr = in + (size_t)row * Dm;

  float v[4], s = 0.f, s2 = 0.f;
#pragma unroll
  for (int i = 0; i < 4; ++i) {
    const int d = tid + i * 256;
    v[i] = xr[d];
    s += v[i];
    s2 += v[i] * v[i];
  }
  __shared__ float r1[256], r2[256];
  r1[tid] = s; r2[tid] = s2;
  __syncthreads();
  for (int off = 128; off > 0; off >>= 1) {
    if (tid < off) { r1[tid] += r1[tid + off]; r2[tid] += r2[tid + off]; }
    __syncthreads();
  }
  const float mean = r1[0] * (1.0f / Dm);
  const float var  = r2[0] * (1.0f / Dm) - mean * mean;
  const float rstd = rsqrtf(var + 1e-5f);
#pragma unroll
  for (int i = 0; i < 4; ++i) {
    const int d = tid + i * 256;
    const float y = (v[i] - mean) * rstd * g[d] + b[d];
    if constexpr (OBF)
      ((__bf16*)out)[(size_t)row * Dm + d] = (__bf16)y;
    else
      ((float*)out)[(size_t)row * Dm + d] = y;
  }
}

__global__ __launch_bounds__(256) void init_lat(
    const float* __restrict__ latents, float* __restrict__ lat,
    int total, int nd)
{
  const int i = blockIdx.x * blockDim.x + threadIdx.x;
  if (i < total) lat[i] = latents[i % nd];
}

// =====================================================================
// attention: block per (16-query tile, head, bt); WMMA QK^T and PV.
// K row-major [tok][INNER]; V pre-transposed [d][TOK].
// =====================================================================
__global__ __launch_bounds__(256) void attn_kernel(
    const __bf16* __restrict__ qbuf, const __bf16* __restrict__ kbuf,
    const __bf16* __restrict__ vtbuf, __bf16* __restrict__ obuf)
{
  __shared__ __bf16 Qs[16][64];
  __shared__ __bf16 Ss[16][TOK];
  __shared__ float  red[16][16];

  const int q0   = blockIdx.x * 16;
  const int h    = blockIdx.y;
  const int bt   = blockIdx.z;
  const int tid  = threadIdx.x;
  const int wave = __builtin_amdgcn_readfirstlane(tid >> 5);
  const int lane = tid & 31;

  const __bf16* qb = qbuf + (size_t)bt * NLAT * INNER;
  const __bf16* kb = kbuf + (size_t)bt * TOK * INNER;
  const __bf16* vt = vtbuf + (size_t)bt * INNER * TOK;

#if HAVE_ASYNC
  if (tid < 128) {  // 16 rows x 64 cols = 128 x 16B chunks
    const int qi = tid >> 3, c = (tid & 7) * 8;
    async_cp16(qb + (size_t)(q0 + qi) * INNER + h * DH + c, &Qs[qi][c]);
  }
  __builtin_amdgcn_s_wait_asynccnt(0);
#else
#pragma unroll
  for (int i = 0; i < 4; ++i) {
    const int e  = tid + i * 256;
    const int qi = e >> 6, d = e & 63;
    Qs[qi][d] = qb[(size_t)(q0 + qi) * INNER + h * DH + d];
  }
#endif
  __syncthreads();

  // ---- phase 1: sim = q @ k^T ----
  for (int nsub = wave; nsub < TOK / 16; nsub += 8) {
    v8f acc = {0.f, 0.f, 0.f, 0.f, 0.f, 0.f, 0.f, 0.f};
    const int tokn = nsub * 16 + (lane & 15);
#pragma unroll
    for (int kc = 0; kc < DH; kc += 32) {
      FragU a, b;
      const int aoff = (lane < 16) ? 0 : 16;
      a.c[0] = *reinterpret_cast<const v8bf*>(&Qs[lane & 15][kc + aoff]);
      a.c[1] = *reinterpret_cast<const v8bf*>(&Qs[lane & 15][kc + aoff + 8]);
      const __bf16* kp =
          kb + (size_t)tokn * INNER + h * DH + kc + ((lane < 16) ? 0 : 16);
      b.c[0] = *reinterpret_cast<const v8bf*>(kp);
      b.c[1] = *reinterpret_cast<const v8bf*>(kp + 8);
      acc = __builtin_amdgcn_wmma_f32_16x16x32_bf16(false, a.v, false, b.v,
                                                    (short)0, acc, false,
                                                    false);
    }
#pragma unroll
    for (int r = 0; r < 8; ++r) {
      const int rr = r + ((lane < 16) ? 0 : 8);
      Ss[rr][nsub * 16 + (lane & 15)] = (__bf16)acc[r];
    }
  }
  __syncthreads();

  // ---- phase 2: softmax, vectorized 4-wide (17 x 64 = 1088) ----
  {
    const int row = tid >> 4, sub = tid & 15;
    float mx = -3.0e38f;
#pragma unroll
    for (int jj = 0; jj < 17; ++jj) {
      const int c = jj * 64 + sub * 4;
      const v4bf t = *reinterpret_cast<const v4bf*>(&Ss[row][c]);
      mx = fmaxf(mx, fmaxf(fmaxf((float)t[0], (float)t[1]),
                           fmaxf((float)t[2], (float)t[3])));
    }
    red[row][sub] = mx;
    __syncthreads();
    float rowmax = red[row][0];
#pragma unroll
    for (int i = 1; i < 16; ++i) rowmax = fmaxf(rowmax, red[row][i]);
    __syncthreads();
    float s = 0.f;
#pragma unroll
    for (int jj = 0; jj < 17; ++jj) {
      const int c = jj * 64 + sub * 4;
      v4bf t = *reinterpret_cast<v4bf*>(&Ss[row][c]);
      const float p0 = __expf((float)t[0] - rowmax);
      const float p1 = __expf((float)t[1] - rowmax);
      const float p2 = __expf((float)t[2] - rowmax);
      const float p3 = __expf((float)t[3] - rowmax);
      s += p0 + p1 + p2 + p3;
      t[0] = (__bf16)p0; t[1] = (__bf16)p1;
      t[2] = (__bf16)p2; t[3] = (__bf16)p3;
      *reinterpret_cast<v4bf*>(&Ss[row][c]) = t;
    }
    red[row][sub] = s;
    __syncthreads();
    float rsum = 0.f;
#pragma unroll
    for (int i = 0; i < 16; ++i) rsum += red[row][i];
    const float rinv = 1.0f / rsum;
#pragma unroll
    for (int jj = 0; jj < 17; ++jj) {
      const int c = jj * 64 + sub * 4;
      v4bf t = *reinterpret_cast<v4bf*>(&Ss[row][c]);
      t[0] = (__bf16)((float)t[0] * rinv);
      t[1] = (__bf16)((float)t[1] * rinv);
      t[2] = (__bf16)((float)t[2] * rinv);
      t[3] = (__bf16)((float)t[3] * rinv);
      *reinterpret_cast<v4bf*>(&Ss[row][c]) = t;
    }
  }
  __syncthreads();

  // ---- phase 3: o = P @ v  (V transposed: contiguous B fragments) ----
  if (wave < 4) {
    const int nb = wave;
    v8f acc = {0.f, 0.f, 0.f, 0.f, 0.f, 0.f, 0.f, 0.f};
    const int dcol = nb * 16 + (lane & 15);
    const __bf16* vrow = vt + (size_t)(h * DH + dcol) * TOK;
    for (int kc = 0; kc < TOK; kc += 32) {
      FragU a, b;
      const int aoff = (lane < 16) ? 0 : 16;
      a.c[0] = *reinterpret_cast<const v8bf*>(&Ss[lane & 15][kc + aoff]);
      a.c[1] = *reinterpret_cast<const v8bf*>(&Ss[lane & 15][kc + aoff + 8]);
      const int t0 = kc + ((lane < 16) ? 0 : 16);
      b.c[0] = *reinterpret_cast<const v8bf*>(vrow + t0);
      b.c[1] = *reinterpret_cast<const v8bf*>(vrow + t0 + 8);
      acc = __builtin_amdgcn_wmma_f32_16x16x32_bf16(false, a.v, false, b.v,
                                                    (short)0, acc, false,
                                                    false);
    }
#pragma unroll
    for (int r = 0; r < 8; ++r) {
      const int rr = r + ((lane < 16) ? 0 : 8);
      obuf[(size_t)bt * NLAT * INNER + (size_t)(q0 + rr) * INNER + h * DH +
           nb * 16 + (lane & 15)] = (__bf16)acc[r];
    }
  }
}

// =====================================================================
// host driver
// =====================================================================
extern "C" void kernel_launch(void* const* d_in, const int* in_sizes, int n_in,
                              void* d_out, int out_size, void* d_ws,
                              size_t ws_size, hipStream_t stream)
{
  (void)in_sizes; (void)n_in; (void)out_size; (void)ws_size;

  const float* x       = (const float*)d_in[0];
  const float* latents = (const float*)d_in[1];
  const float* femb    = (const float*)d_in[2];
  const float* g_media = (const float*)d_in[3];
  const float* b_media = (const float*)d_in[4];
  const float* g_lat   = (const float*)d_in[5];
  const float* b_lat   = (const float*)d_in[6];
  const float* Wq      = (const float*)d_in[7];
  const float* Wkv     = (const float*)d_in[8];
  const float* Wo      = (const float*)d_in[9];
  const float* g_ff    = (const float*)d_in[10];
  const float* b_ff    = (const float*)d_in[11];
  const float* W1      = (const float*)d_in[12];
  const float* W2      = (const float*)d_in[13];
  const float* g_out   = (const float*)d_in[14];
  const float* b_out   = (const float*)d_in[15];

  char* ws = (char*)d_ws;
  size_t off = 0;
  auto alloc = [&](size_t bytes) {
    char* p = ws + off;
    off = (off + bytes + 255) & ~(size_t)255;
    return p;
  };
  __bf16* xm  = (__bf16*)alloc((size_t)BT * MEDIA * Dm * 2);
  float*  lat = (float*)alloc((size_t)BT * NLAT * Dm * 4);
  __bf16* lnl = (__bf16*)alloc((size_t)BT * NLAT * Dm * 2);
  __bf16* qb  = (__bf16*)alloc((size_t)BT * NLAT * INNER * 2);
  __bf16* kb  = (__bf16*)alloc((size_t)BT * TOK * INNER * 2);
  __bf16* vtb = (__bf16*)alloc((size_t)BT * INNER * TOK * 2);
  __bf16* ob  = (__bf16*)alloc((size_t)BT * NLAT * INNER * 2);
  __bf16* hb  = (__bf16*)alloc((size_t)BT * NLAT * FFD * 2);
  // bf16 transposed weights [N][K]
  __bf16* Wq_t  = (__bf16*)alloc((size_t)Lnum * INNER * Dm * 2);
  __bf16* Wkv_t = (__bf16*)alloc((size_t)Lnum * KV2 * Dm * 2);
  __bf16* Wo_t  = (__bf16*)alloc((size_t)Lnum * Dm * INNER * 2);
  __bf16* W1_t  = (__bf16*)alloc((size_t)Lnum * FFD * Dm * 2);
  __bf16* W2_t  = (__bf16*)alloc((size_t)Lnum * Dm * FFD * 2);
  __bf16* Wf_t  = (__bf16*)alloc((size_t)KV2 * Dm * 2);  // folded media Wkv^T
  float*  biasf = (float*)alloc((size_t)KV2 * 4);

  auto cvtT = [&](const float* in, __bf16* out, int K, int N) {
    const int nthr = K / 4 * N;
    cvt_bf16_t<<<(nthr + 255) / 256, 256, 0, stream>>>(in, out, K, N);
  };
  for (int l = 0; l < Lnum; ++l) {
    cvtT(Wq + (size_t)l * Dm * INNER, Wq_t + (size_t)l * INNER * Dm, Dm, INNER);
    cvtT(Wkv + (size_t)l * Dm * KV2, Wkv_t + (size_t)l * KV2 * Dm, Dm, KV2);
    cvtT(Wo + (size_t)l * INNER * Dm, Wo_t + (size_t)l * Dm * INNER, INNER, Dm);
    cvtT(W1 + (size_t)l * Dm * FFD, W1_t + (size_t)l * FFD * Dm, Dm, FFD);
    cvtT(W2 + (size_t)l * FFD * Dm, W2_t + (size_t)l * Dm * FFD, FFD, Dm);
  }

  prep_media<<<BT * MEDIA, 256, 0, stream>>>(x, femb, xm);
  {
    const int total = BT * NLAT * Dm;
    init_lat<<<(total + 255) / 256, 256, 0, stream>>>(latents, lat, total,
                                                      NLAT * Dm);
  }

  const float qscale = 0.125f;  // DH^-0.5
  const size_t kvBatch = (size_t)TOK * INNER;  // per-bt stride for K and Vt

  for (int l = 0; l < Lnum; ++l) {
    const float* gm    = g_media + (size_t)l * Dm;
    const float* bm    = b_media + (size_t)l * Dm;
    const float* gl    = g_lat + (size_t)l * Dm;
    const float* bl    = b_lat + (size_t)l * Dm;
    const float* gf    = g_ff + (size_t)l * Dm;
    const float* bf    = b_ff + (size_t)l * Dm;
    const float* Wkv_l = Wkv + (size_t)l * Dm * KV2;
    const __bf16* Wq_l  = Wq_t + (size_t)l * INNER * Dm;
    const __bf16* Wkvtl = Wkv_t + (size_t)l * KV2 * Dm;
    const __bf16* Wo_l  = Wo_t + (size_t)l * Dm * INNER;
    const __bf16* W1_l  = W1_t + (size_t)l * FFD * Dm;
    const __bf16* W2_l  = W2_t + (size_t)l * Dm * FFD;

    // fold media LN affine into Wkv^T: Wf_t[n][k] = gm[k]*Wkv[k][n]
    fold_wkv_t<<<KV2 * Dm / 4 / 256, 256, 0, stream>>>(Wkv_l, gm, Wf_t);
    bias_wkv<<<KV2 / 256, 256, 0, stream>>>(Wkv_l, bm, biasf);

    // latent LN (affine, bf16)
    ln_rows<true><<<BT * NLAT, 256, 0, stream>>>(lat, gl, bl, lnl);

    // q = ln_l @ Wq * scale
    gemm_bf16_wmma<0><<<dim3(INNER / 128, 1, BT), 256, 0, stream>>>(
        lnl, Dm, (size_t)NLAT * Dm, Wq_l, Dm, nullptr,
        qb, nullptr, INNER, (size_t)NLAT * INNER, 0, nullptr, 0, 0, Dm,
        qscale);

    // kv latent rows (tokens 1024..1087), split K / V^T
    gemm_bf16_wmma<3><<<dim3(KV2 / 128, 1, BT), 256, 0, stream>>>(
        lnl, Dm, (size_t)NLAT * Dm, Wkvtl, Dm, nullptr,
        kb, vtb, 0, kvBatch, MEDIA, nullptr, 0, 0, Dm, 1.0f);

    // kv media rows (folded affine + bias), split K / V^T -- dominant GEMM
    gemm_bf16_wmma<3><<<dim3(KV2 / 128, MEDIA / 64, BT), 256, 0, stream>>>(
        xm, Dm, (size_t)MEDIA * Dm, Wf_t, Dm, biasf,
        kb, vtb, 0, kvBatch, 0, nullptr, 0, 0, Dm, 1.0f);

    attn_kernel<<<dim3(NLAT / 16, Hh, BT), 256, 0, stream>>>(qb, kb, vtb, ob);

    // lat = o @ Wo + lat
    gemm_bf16_wmma<1><<<dim3(Dm / 128, 1, BT), 256, 0, stream>>>(
        ob, INNER, (size_t)NLAT * INNER, Wo_l, INNER, nullptr,
        lat, nullptr, Dm, (size_t)NLAT * Dm, 0, lat, Dm, (size_t)NLAT * Dm,
        INNER, 1.0f);

    // FF
    ln_rows<true><<<BT * NLAT, 256, 0, stream>>>(lat, gf, bf, lnl);

    gemm_bf16_wmma<2><<<dim3(FFD / 128, 1, BT), 256, 0, stream>>>(
        lnl, Dm, (size_t)NLAT * Dm, W1_l, Dm, nullptr,
        hb, nullptr, FFD, (size_t)NLAT * FFD, 0, nullptr, 0, 0, Dm, 1.0f);

    gemm_bf16_wmma<1><<<dim3(Dm / 128, 1, BT), 256, 0, stream>>>(
        hb, FFD, (size_t)NLAT * FFD, W2_l, FFD, nullptr,
        lat, nullptr, Dm, (size_t)NLAT * Dm, 0, lat, Dm, (size_t)NLAT * Dm,
        FFD, 1.0f);
  }

  ln_rows<false><<<BT * NLAT, 256, 0, stream>>>(lat, g_out, b_out, d_out);
}
